// LongformerClassifier_44598940401809
// MI455X (gfx1250) — matrix operand order repaired and checked by reading, not compiled, not run on hardware
//
#include <hip/hip_runtime.h>
#include <hip/hip_bf16.h>
#include <math.h>

// ---------------- model dims ----------------
#define S_LEN     4096
#define D_MODEL   768
#define NHEAD     12
#define HEAD_DIM  64
#define NLAYER    2
#define FF_DIM    3072
#define WIN_C     256
#define NBLK_T    (S_LEN / WIN_C)   // 16
#define NUM_LABELS 50
#define NEG_SCORE (-1.0e9f)

typedef __attribute__((ext_vector_type(16))) __bf16 v16bf;
typedef __attribute__((ext_vector_type(8)))  float  v8f;

// ---------------- CDNA5 async global->LDS path (guarded) ----------------
#ifndef __has_builtin
#define __has_builtin(x) 0
#endif
#if __has_builtin(__builtin_amdgcn_global_load_async_to_lds_b128)
#define USE_ASYNC_LDS 1
#else
#define USE_ASYNC_LDS 0
#endif

// The builtin expects pointers to 16-byte int vectors (gcc vector_size style),
// qualified with global (AS1) / LDS (AS3) address spaces.
typedef int v4i_gcc __attribute__((vector_size(16)));
typedef __attribute__((address_space(1))) v4i_gcc as1_v4i;
typedef __attribute__((address_space(3))) v4i_gcc as3_v4i;

// copy 16 bytes (8 bf16) global -> LDS
__device__ inline void copy16_g2l(const __bf16* g, __bf16* l) {
#if USE_ASYNC_LDS
  __builtin_amdgcn_global_load_async_to_lds_b128((as1_v4i*)g, (as3_v4i*)l, 0, 0);
#else
  *(uint4*)l = *(const uint4*)g;
#endif
}

__device__ inline void async_wait_all() {
#if USE_ASYNC_LDS
#if __has_builtin(__builtin_amdgcn_s_wait_asynccnt)
  __builtin_amdgcn_s_wait_asynccnt(0);
#else
  asm volatile("s_wait_asynccnt 0x0" ::: "memory");
#endif
#endif
}

// ---------------- WMMA helpers (CDNA5 wave32) ----------------
__device__ inline v8f wmma_bf16(v16bf a, v16bf b, v8f c) {
  return __builtin_amdgcn_wmma_f32_16x16x32_bf16(false, a, false, b, (short)0, c, false, false);
}

// A-fragment (16x32 bf16, MxK). Lane l: m=l&15, kbase=(l>>4)*16; one 32B run.
__device__ inline v16bf load_a_frag(const __bf16* A, int row_base, long ld, int k0) {
  int lane = threadIdx.x & 31;
  int m    = lane & 15;
  int kb   = (lane >> 4) * 16;
  const __bf16* p = A + (long)(row_base + m) * ld + k0 + kb;
  v16bf v;
#pragma unroll
  for (int i = 0; i < 16; ++i) v[i] = p[i];
  return v;
}

// B-fragment (32x16 bf16, KxN) from an [N][K]-contiguous buffer.
// Lane l: n=l&15, K = k_off + (l>>4)*16 + e  -> one 32B run.
__device__ inline v16bf load_b_frag_contig(const __bf16* B, int n_base, int ld, int k_off) {
  int lane = threadIdx.x & 31;
  const __bf16* p = B + (long)(n_base + (lane & 15)) * ld + k_off + (lane >> 4) * 16;
  v16bf v;
#pragma unroll
  for (int i = 0; i < 16; ++i) v[i] = p[i];
  return v;
}

// B-fragment where "n" indexes rows of a row-major matrix (Q@K^T). Per-lane row clamp.
__device__ inline v16bf load_b_frag_rows(const __bf16* K, int row_base, long ld, int d0, int max_row) {
  int lane = threadIdx.x & 31;
  int row  = row_base + (lane & 15);
  row = row < 0 ? 0 : (row > max_row ? max_row : row);
  const __bf16* p = K + (long)row * ld + d0 + (lane >> 4) * 16;
  v16bf v;
#pragma unroll
  for (int i = 0; i < 16; ++i) v[i] = p[i];
  return v;
}

__device__ inline float gelu_tanh(float x) {
  float x3 = x * x * x;
  return 0.5f * x * (1.0f + tanhf(0.7978845608028654f * (x + 0.044715f * x3)));
}

// stage a 64-row x 32-col bf16 tile (src row-major, ld_src) into LDS [64][32]
__device__ inline void stage_tile_64x32(const __bf16* src, long ld_src, __bf16* dst, int tid) {
#pragma unroll
  for (int s = tid; s < 256; s += 128) {
    int r = s >> 2, ko = (s & 3) * 8;
    copy16_g2l(src + (long)r * ld_src + ko, dst + r * 32 + ko);
  }
}

// ---------------- generic bf16 GEMM:  out = act((A@Wt^T + bias) * alpha) ----------------
// A: MxK bf16 row-major. Wt: NxK bf16 row-major (pre-transposed weights).
// Block = 64x64 tile, 128 threads (4 waves), double-buffered async LDS staging.
__global__ __launch_bounds__(128)
void gemm_bf16_kernel(const __bf16* __restrict__ A, const __bf16* __restrict__ Wt,
                      const float* __restrict__ bias, float alpha, int apply_gelu,
                      float* __restrict__ outF, __bf16* __restrict__ outB,
                      int M, int K, int N) {
  __shared__ __align__(16) __bf16 As[2][64 * 32];
  __shared__ __align__(16) __bf16 Bs[2][64 * 32];
  const int bn = blockIdx.x * 64;
  const int bm = blockIdx.y * 64;
  const int tid  = threadIdx.x;
  const int lane = tid & 31;
  const int wave = tid >> 5;
  const int wm = (wave >> 1) * 32;
  const int wn = (wave & 1) * 32;

  v8f acc[2][2];
#pragma unroll
  for (int i = 0; i < 2; ++i)
#pragma unroll
    for (int j = 0; j < 2; ++j)
#pragma unroll
      for (int r = 0; r < 8; ++r) acc[i][j][r] = 0.0f;

  stage_tile_64x32(A  + (long)bm * K, K, &As[0][0], tid);
  stage_tile_64x32(Wt + (long)bn * K, K, &Bs[0][0], tid);
  async_wait_all();
  __syncthreads();

  int cur = 0;
  for (int k0 = 0; k0 < K; k0 += 32) {
    if (k0 + 32 < K) {
      stage_tile_64x32(A  + (long)bm * K + k0 + 32, K, &As[cur ^ 1][0], tid);
      stage_tile_64x32(Wt + (long)bn * K + k0 + 32, K, &Bs[cur ^ 1][0], tid);
      if (k0 + 64 < K) __builtin_prefetch(A + (long)bm * K + k0 + 64, 0, 0);
    }

    v16bf af[2], bf[2];
#pragma unroll
    for (int mt = 0; mt < 2; ++mt) af[mt] = load_a_frag(&As[cur][0], wm + mt * 16, 32, 0);
#pragma unroll
    for (int nt = 0; nt < 2; ++nt) bf[nt] = load_b_frag_contig(&Bs[cur][0], wn + nt * 16, 32, 0);
#pragma unroll
    for (int mt = 0; mt < 2; ++mt)
#pragma unroll
      for (int nt = 0; nt < 2; ++nt) acc[mt][nt] = wmma_bf16(af[mt], bf[nt], acc[mt][nt]);

    async_wait_all();
    __syncthreads();
    cur ^= 1;
  }

  const int g = lane >> 4, cn = lane & 15;
#pragma unroll
  for (int mt = 0; mt < 2; ++mt)
#pragma unroll
    for (int nt = 0; nt < 2; ++nt)
#pragma unroll
      for (int r = 0; r < 8; ++r) {
        int row = bm + wm + mt * 16 + g * 8 + r;
        int col = bn + wn + nt * 16 + cn;
        float v = acc[mt][nt][r];
        if (bias) v += bias[col];
        v *= alpha;
        if (apply_gelu) v = gelu_tanh(v);
        if (outF) outF[(long)row * N + col] = v;
        if (outB) outB[(long)row * N + col] = (__bf16)v;
      }
}

// ---------------- sliding-window + global-key attention (flash style) ----------------
__global__ __launch_bounds__(256)
void sliding_attn_kernel(const __bf16* __restrict__ q, const __bf16* __restrict__ k,
                         const __bf16* __restrict__ v, const int* __restrict__ amask,
                         __bf16* __restrict__ attn) {
  const int t = blockIdx.x, h = blockIdx.y;
  const int tid = threadIdx.x, lane = tid & 31, wave = tid >> 5;
  const int qbase = wave * 32;
  const int g = lane >> 4, cn = lane & 15;
  const int dh = h * HEAD_DIM;
  const int win0 = (t - 1) * WIN_C;

  __shared__ __align__(16) __bf16 Pst[8][32][64];
  __shared__ __align__(16) __bf16 Vrm[64][64];
  __shared__ __align__(16) __bf16 Vt[64][64];
  __shared__ float sgs[256];

  v16bf aq[2][2];
#pragma unroll
  for (int mt = 0; mt < 2; ++mt)
#pragma unroll
    for (int ks = 0; ks < 2; ++ks)
      aq[mt][ks] = load_a_frag(q, t * WIN_C + qbase + mt * 16, D_MODEL, dh + ks * 32);

  v8f o[2][4];
  float mrow[2][8], lrow[2][8];
#pragma unroll
  for (int mt = 0; mt < 2; ++mt) {
#pragma unroll
    for (int nt = 0; nt < 4; ++nt)
#pragma unroll
      for (int r = 0; r < 8; ++r) o[mt][nt][r] = 0.0f;
#pragma unroll
    for (int r = 0; r < 8; ++r) { mrow[mt][r] = -1.0e30f; lrow[mt][r] = 0.0f; }
  }

  for (int ch = 0; ch < 12; ++ch) {
    const int jb = ch * 64;

    // kick off async staging of this chunk's V rows (row-major) while we do Q@K^T
    __syncthreads();
#pragma unroll
    for (int s2 = tid; s2 < 512; s2 += 256) {
      int row = s2 >> 3, ko = (s2 & 7) * 8;
      int kpos = win0 + jb + row;
      int rr = kpos < 0 ? 0 : (kpos >= S_LEN ? S_LEN - 1 : kpos);
      copy16_g2l(v + (long)rr * D_MODEL + dh + ko, &Vrm[row][ko]);
    }

    v8f sc[2][4];
#pragma unroll
    for (int mt = 0; mt < 2; ++mt)
#pragma unroll
      for (int nt = 0; nt < 4; ++nt)
#pragma unroll
        for (int r = 0; r < 8; ++r) sc[mt][nt][r] = 0.0f;

    // S = Q @ K^T over this 64-key chunk
#pragma unroll
    for (int ks = 0; ks < 2; ++ks)
#pragma unroll
      for (int nt = 0; nt < 4; ++nt) {
        v16bf bk = load_b_frag_rows(k, win0 + jb + nt * 16, D_MODEL, dh + ks * 32, S_LEN - 1);
#pragma unroll
        for (int mt = 0; mt < 2; ++mt) sc[mt][nt] = wmma_bf16(aq[mt][ks], bk, sc[mt][nt]);
      }

    // mask
#pragma unroll
    for (int nt = 0; nt < 4; ++nt) {
      int j = jb + nt * 16 + cn;
      int kpos = win0 + j;
      int kc = kpos < 0 ? 0 : (kpos >= S_LEN ? S_LEN - 1 : kpos);
      bool colok = (kpos >= 0) && (kpos < S_LEN) && (kpos != 0) && (amask[kc] > 0);
#pragma unroll
      for (int mt = 0; mt < 2; ++mt)
#pragma unroll
        for (int r = 0; r < 8; ++r) {
          int qlocal = qbase + mt * 16 + g * 8 + r;
          int rel = j - WIN_C - qlocal;
          if (!(colok && rel >= -WIN_C && rel <= WIN_C)) sc[mt][nt][r] = NEG_SCORE;
        }
    }

    // online softmax (wave32 16-lane row reductions)
#pragma unroll
    for (int mt = 0; mt < 2; ++mt)
#pragma unroll
      for (int r = 0; r < 8; ++r) {
        float rm = NEG_SCORE;
#pragma unroll
        for (int nt = 0; nt < 4; ++nt) rm = fmaxf(rm, sc[mt][nt][r]);
        for (int off = 1; off < 16; off <<= 1) rm = fmaxf(rm, __shfl_xor(rm, off, 32));
        float m_old = mrow[mt][r];
        float m_new = fmaxf(m_old, rm);
        float scale = __expf(m_old - m_new);
        float rs = 0.0f;
#pragma unroll
        for (int nt = 0; nt < 4; ++nt) {
          float s = sc[mt][nt][r];
          float p = (s <= -1.0e8f) ? 0.0f : __expf(s - m_new);
          sc[mt][nt][r] = p;
          rs += p;
        }
        for (int off = 1; off < 16; off <<= 1) rs += __shfl_xor(rs, off, 32);
        mrow[mt][r] = m_new;
        lrow[mt][r] = lrow[mt][r] * scale + rs;
#pragma unroll
        for (int nt = 0; nt < 4; ++nt) o[mt][nt][r] *= scale;
      }

    // store P (bf16) into per-wave LDS region
#pragma unroll
    for (int mt = 0; mt < 2; ++mt)
#pragma unroll
      for (int nt = 0; nt < 4; ++nt)
#pragma unroll
        for (int r = 0; r < 8; ++r)
          Pst[wave][mt * 16 + g * 8 + r][nt * 16 + cn] = (__bf16)sc[mt][nt][r];

    // finish async V staging, transpose inside LDS
    async_wait_all();
    __syncthreads();
    for (int i = tid; i < 64 * 64; i += 256) {
      int kl = i & 63, hd = i >> 6;
      Vt[hd][kl] = Vrm[kl][hd];
    }
    __syncthreads();

    // O += P @ V
#pragma unroll
    for (int ks = 0; ks < 2; ++ks)
#pragma unroll
      for (int mt = 0; mt < 2; ++mt) {
        v16bf ap = load_a_frag(&Pst[wave][0][0], mt * 16, 64, ks * 32);
#pragma unroll
        for (int nt = 0; nt < 4; ++nt) {
          v16bf bv = load_b_frag_contig(&Vt[0][0], nt * 16, 64, ks * 32);
          o[mt][nt] = wmma_bf16(ap, bv, o[mt][nt]);
        }
      }
  }

  // global key (kpos==0): s_g = q_row . k0
  __syncthreads();
  {
    float s = 0.0f;
    const long qoff = (long)(t * WIN_C + tid) * D_MODEL + dh;
    for (int d0 = 0; d0 < HEAD_DIM; ++d0) s += (float)q[qoff + d0] * (float)k[dh + d0];
    sgs[tid] = s;
  }
  __syncthreads();

  float v0[4];
#pragma unroll
  for (int nt = 0; nt < 4; ++nt) v0[nt] = (float)v[dh + nt * 16 + cn];

#pragma unroll
  for (int mt = 0; mt < 2; ++mt)
#pragma unroll
    for (int r = 0; r < 8; ++r) {
      int rl = qbase + mt * 16 + g * 8 + r;
      float sg = sgs[rl];
      float m = mrow[mt][r], l = lrow[mt][r];
      float M = fmaxf(m, sg);
      float eM = __expf(m - M), eG = __expf(sg - M);
      float inv = 1.0f / (l * eM + eG);
      long row = (long)(t * WIN_C + rl);
#pragma unroll
      for (int nt = 0; nt < 4; ++nt) {
        float ov = (o[mt][nt][r] * eM + eG * v0[nt]) * inv;
        attn[row * D_MODEL + dh + nt * 16 + cn] = (__bf16)ov;
      }
    }
}

// ---------------- row-0 global attention (per head) ----------------
__global__ __launch_bounds__(256)
void global_attn_kernel(const float* __restrict__ x, const float* __restrict__ Wqg,
                        const float* __restrict__ bqg, const __bf16* __restrict__ kg,
                        const __bf16* __restrict__ vg, const int* __restrict__ amask,
                        __bf16* __restrict__ attn) {
  const int h = blockIdx.x, tid = threadIdx.x, dh = h * HEAD_DIM;
  __shared__ float qg[HEAD_DIM];
  __shared__ float sc[S_LEN];
  __shared__ float red[256];
  __shared__ float op[HEAD_DIM][4];

  if (tid < HEAD_DIM) {
    float a = bqg[dh + tid];
    for (int d = 0; d < D_MODEL; ++d) a += x[d] * Wqg[(long)d * D_MODEL + dh + tid];
    qg[tid] = a * 0.125f;
  }
  __syncthreads();

  for (int s = tid; s < S_LEN; s += 256) {
    float a = 0.0f;
    for (int d = 0; d < HEAD_DIM; ++d) a += qg[d] * (float)kg[(long)s * D_MODEL + dh + d];
    sc[s] = (amask[s] > 0) ? a : NEG_SCORE;
  }
  __syncthreads();

  float mx = -1.0e30f;
  for (int s = tid; s < S_LEN; s += 256) mx = fmaxf(mx, sc[s]);
  red[tid] = mx; __syncthreads();
  for (int st = 128; st > 0; st >>= 1) { if (tid < st) red[tid] = fmaxf(red[tid], red[tid + st]); __syncthreads(); }
  mx = red[0]; __syncthreads();

  float sm = 0.0f;
  for (int s = tid; s < S_LEN; s += 256) { float e = __expf(sc[s] - mx); sc[s] = e; sm += e; }
  red[tid] = sm; __syncthreads();
  for (int st = 128; st > 0; st >>= 1) { if (tid < st) red[tid] += red[tid + st]; __syncthreads(); }
  float inv = 1.0f / red[0];
  __syncthreads();

  int hd = tid >> 2, part = tid & 3;
  float acc = 0.0f;
  for (int s = part; s < S_LEN; s += 4) acc += sc[s] * (float)vg[(long)s * D_MODEL + dh + hd];
  op[hd][part] = acc;
  __syncthreads();
  if (tid < HEAD_DIM) {
    float o = (op[tid][0] + op[tid][1] + op[tid][2] + op[tid][3]) * inv;
    attn[dh + tid] = (__bf16)o;
  }
}

// ---------------- embed + LN ----------------
__global__ __launch_bounds__(256)
void embed_ln_kernel(const int* __restrict__ ids, const float* __restrict__ tok,
                     const float* __restrict__ pos, const float* __restrict__ gamma,
                     const float* __restrict__ beta, float* __restrict__ xf,
                     __bf16* __restrict__ xb) {
  const int s = blockIdx.x, tid = threadIdx.x;
  __shared__ float buf[D_MODEL];
  __shared__ float red[256];
  const long tb = (long)ids[s] * D_MODEL;
  float ps = 0.0f;
  for (int d = tid; d < D_MODEL; d += 256) { float e = tok[tb + d] + pos[(long)s * D_MODEL + d]; buf[d] = e; ps += e; }
  red[tid] = ps; __syncthreads();
  for (int st = 128; st > 0; st >>= 1) { if (tid < st) red[tid] += red[tid + st]; __syncthreads(); }
  float mean = red[0] / D_MODEL; __syncthreads();
  float pv = 0.0f;
  for (int d = tid; d < D_MODEL; d += 256) { float t2 = buf[d] - mean; pv += t2 * t2; }
  red[tid] = pv; __syncthreads();
  for (int st = 128; st > 0; st >>= 1) { if (tid < st) red[tid] += red[tid + st]; __syncthreads(); }
  float rstd = rsqrtf(red[0] / D_MODEL + 1e-5f);
  for (int d = tid; d < D_MODEL; d += 256) {
    float y = (buf[d] - mean) * rstd * gamma[d] + beta[d];
    xf[(long)s * D_MODEL + d] = y;
    xb[(long)s * D_MODEL + d] = (__bf16)y;
  }
}

// ---------------- residual + LN ----------------
__global__ __launch_bounds__(256)
void resid_ln_kernel(const float* __restrict__ x, const float* __restrict__ a,
                     const float* __restrict__ gamma, const float* __restrict__ beta,
                     float* __restrict__ yf, __bf16* __restrict__ yb) {
  const int s = blockIdx.x, tid = threadIdx.x;
  __shared__ float buf[D_MODEL];
  __shared__ float red[256];
  float ps = 0.0f;
  for (int d = tid; d < D_MODEL; d += 256) {
    float e = x[(long)s * D_MODEL + d] + a[(long)s * D_MODEL + d];
    buf[d] = e; ps += e;
  }
  red[tid] = ps; __syncthreads();
  for (int st = 128; st > 0; st >>= 1) { if (tid < st) red[tid] += red[tid + st]; __syncthreads(); }
  float mean = red[0] / D_MODEL; __syncthreads();
  float pv = 0.0f;
  for (int d = tid; d < D_MODEL; d += 256) { float t2 = buf[d] - mean; pv += t2 * t2; }
  red[tid] = pv; __syncthreads();
  for (int st = 128; st > 0; st >>= 1) { if (tid < st) red[tid] += red[tid + st]; __syncthreads(); }
  float rstd = rsqrtf(red[0] / D_MODEL + 1e-5f);
  for (int d = tid; d < D_MODEL; d += 256) {
    float y = (buf[d] - mean) * rstd * gamma[d] + beta[d];
    yf[(long)s * D_MODEL + d] = y;
    yb[(long)s * D_MODEL + d] = (__bf16)y;
  }
}

// ---------------- weight transpose+convert: W[K][N] f32 -> Wt[N][K] bf16 ----------------
__global__ __launch_bounds__(256)
void transpose_to_bf16_kernel(const float* __restrict__ W, __bf16* __restrict__ Wt,
                              int K, int N) {
  __shared__ float tile[32][33];
  const int kb = blockIdx.y * 32, nb = blockIdx.x * 32;
  const int tx = threadIdx.x & 31, ty = threadIdx.x >> 5;
  for (int i = ty; i < 32; i += 8) tile[i][tx] = W[(long)(kb + i) * N + nb + tx];
  __syncthreads();
  for (int i = ty; i < 32; i += 8) Wt[(long)(nb + i) * K + kb + tx] = (__bf16)tile[tx][i];
}

// ---------------- classifier ----------------
__global__ void classifier_kernel(const float* __restrict__ x0, const float* __restrict__ W,
                                  const float* __restrict__ b, float* __restrict__ out) {
  int o = threadIdx.x;
  if (o < NUM_LABELS) {
    float a = b[o];
    for (int d = 0; d < D_MODEL; ++d) a += x0[d] * W[(long)d * NUM_LABELS + o];
    out[o] = a;
  }
}

// ---------------- host-side launch ----------------
extern "C" void kernel_launch(void* const* d_in, const int* in_sizes, int n_in,
                              void* d_out, int out_size, void* d_ws, size_t ws_size,
                              hipStream_t stream) {
  (void)in_sizes; (void)n_in; (void)out_size; (void)ws_size;
  const int*   ids   = (const int*)d_in[0];
  const int*   amask = (const int*)d_in[1];
  const float* clfW  = (const float*)d_in[2];
  const float* clfB  = (const float*)d_in[3];
  const float* embG  = (const float*)d_in[4];
  const float* embB  = (const float*)d_in[5];
  const float* posE  = (const float*)d_in[6 + NLAYER * 22];
  const float* tokE  = (const float*)d_in[7 + NLAYER * 22];

  struct LayerP {
    const float *W1, *W2, *Wk, *Wkg, *Wo, *Wq, *Wqg, *Wv, *Wvg;
    const float *b1, *b2, *bk, *bkg, *bo, *bq, *bqg, *bv, *bvg;
    const float *ln1g, *ln1b, *ln2g, *ln2b;
  } L[NLAYER];
  for (int l = 0; l < NLAYER; ++l) {
    int b = 6 + l * 22;
    L[l].W1  = (const float*)d_in[b + 0];  L[l].W2  = (const float*)d_in[b + 1];
    L[l].Wk  = (const float*)d_in[b + 2];  L[l].Wkg = (const float*)d_in[b + 3];
    L[l].Wo  = (const float*)d_in[b + 4];  L[l].Wq  = (const float*)d_in[b + 5];
    L[l].Wqg = (const float*)d_in[b + 6];  L[l].Wv  = (const float*)d_in[b + 7];
    L[l].Wvg = (const float*)d_in[b + 8];
    L[l].b1  = (const float*)d_in[b + 9];  L[l].b2  = (const float*)d_in[b + 10];
    L[l].bk  = (const float*)d_in[b + 11]; L[l].bkg = (const float*)d_in[b + 12];
    L[l].bo  = (const float*)d_in[b + 13]; L[l].bq  = (const float*)d_in[b + 14];
    L[l].bqg = (const float*)d_in[b + 15]; L[l].bv  = (const float*)d_in[b + 16];
    L[l].bvg = (const float*)d_in[b + 17];
    L[l].ln1g = (const float*)d_in[b + 18]; L[l].ln1b = (const float*)d_in[b + 19];
    L[l].ln2g = (const float*)d_in[b + 20]; L[l].ln2b = (const float*)d_in[b + 21];
  }

  char* wsp = (char*)d_ws;
  auto bump = [&](size_t bytes) -> char* {
    char* p = wsp; wsp += (bytes + 255) & ~(size_t)255; return p;
  };
  const long DD = (long)D_MODEL * D_MODEL;
  const long DF = (long)D_MODEL * FF_DIM;
  const long SD = (long)S_LEN * D_MODEL;

  __bf16 *Wqb[NLAYER], *Wkb[NLAYER], *Wvb[NLAYER], *Wob[NLAYER],
         *Wkgb[NLAYER], *Wvgb[NLAYER], *W1b[NLAYER], *W2b[NLAYER];
  for (int l = 0; l < NLAYER; ++l) {
    Wqb[l]  = (__bf16*)bump(DD * 2); Wkb[l]  = (__bf16*)bump(DD * 2);
    Wvb[l]  = (__bf16*)bump(DD * 2); Wob[l]  = (__bf16*)bump(DD * 2);
    Wkgb[l] = (__bf16*)bump(DD * 2); Wvgb[l] = (__bf16*)bump(DD * 2);
    W1b[l]  = (__bf16*)bump(DF * 2); W2b[l]  = (__bf16*)bump(DF * 2);
  }
  float*  xf    = (float*)bump(SD * 4);
  __bf16* xb    = (__bf16*)bump(SD * 2);
  __bf16* qb    = (__bf16*)bump(SD * 2);
  __bf16* kb    = (__bf16*)bump(SD * 2);
  __bf16* vb    = (__bf16*)bump(SD * 2);
  __bf16* kgb   = (__bf16*)bump(SD * 2);
  __bf16* vgb   = (__bf16*)bump(SD * 2);
  __bf16* attnb = (__bf16*)bump(SD * 2);
  float*  af    = (float*)bump(SD * 4);
  __bf16* hb    = (__bf16*)bump((long)S_LEN * FF_DIM * 2);

  auto tconv = [&](const float* s, __bf16* d, int K, int N) {
    dim3 grid(N / 32, K / 32);
    transpose_to_bf16_kernel<<<grid, 256, 0, stream>>>(s, d, K, N);
  };
  for (int l = 0; l < NLAYER; ++l) {
    tconv(L[l].Wq,  Wqb[l],  D_MODEL, D_MODEL);
    tconv(L[l].Wk,  Wkb[l],  D_MODEL, D_MODEL);
    tconv(L[l].Wv,  Wvb[l],  D_MODEL, D_MODEL);
    tconv(L[l].Wo,  Wob[l],  D_MODEL, D_MODEL);
    tconv(L[l].Wkg, Wkgb[l], D_MODEL, D_MODEL);
    tconv(L[l].Wvg, Wvgb[l], D_MODEL, D_MODEL);
    tconv(L[l].W1,  W1b[l],  D_MODEL, FF_DIM);
    tconv(L[l].W2,  W2b[l],  FF_DIM,  D_MODEL);
  }

  embed_ln_kernel<<<S_LEN, 256, 0, stream>>>(ids, tokE, posE, embG, embB, xf, xb);

  auto gemm = [&](const __bf16* A, const __bf16* Wt, const float* bias, float alpha,
                  int gelu, float* oF, __bf16* oB, int M, int K, int N) {
    dim3 grid(N / 64, M / 64);
    gemm_bf16_kernel<<<grid, 128, 0, stream>>>(A, Wt, bias, alpha, gelu, oF, oB, M, K, N);
  };

  for (int l = 0; l < NLAYER; ++l) {
    gemm(xb, Wqb[l],  L[l].bq,  0.125f, 0, nullptr, qb,  S_LEN, D_MODEL, D_MODEL);
    gemm(xb, Wkb[l],  L[l].bk,  1.0f,   0, nullptr, kb,  S_LEN, D_MODEL, D_MODEL);
    gemm(xb, Wvb[l],  L[l].bv,  1.0f,   0, nullptr, vb,  S_LEN, D_MODEL, D_MODEL);
    gemm(xb, Wkgb[l], L[l].bkg, 1.0f,   0, nullptr, kgb, S_LEN, D_MODEL, D_MODEL);
    gemm(xb, Wvgb[l], L[l].bvg, 1.0f,   0, nullptr, vgb, S_LEN, D_MODEL, D_MODEL);

    sliding_attn_kernel<<<dim3(NBLK_T, NHEAD), 256, 0, stream>>>(qb, kb, vb, amask, attnb);
    global_attn_kernel<<<NHEAD, 256, 0, stream>>>(xf, L[l].Wqg, L[l].bqg, kgb, vgb, amask, attnb);

    gemm(attnb, Wob[l], L[l].bo, 1.0f, 0, af, nullptr, S_LEN, D_MODEL, D_MODEL);
    resid_ln_kernel<<<S_LEN, 256, 0, stream>>>(xf, af, L[l].ln1g, L[l].ln1b, xf, xb);

    gemm(xb, W1b[l], L[l].b1, 1.0f, 1, nullptr, hb, S_LEN, D_MODEL, FF_DIM);
    gemm(hb, W2b[l], L[l].b2, 1.0f, 0, af, nullptr, S_LEN, FF_DIM, D_MODEL);
    resid_ln_kernel<<<S_LEN, 256, 0, stream>>>(xf, af, L[l].ln2g, L[l].ln2b, xf, xb);
  }

  classifier_kernel<<<1, 64, 0, stream>>>(xf, clfW, clfB, (float*)d_out);
}
